// CausalSelfAttention_39865886441606
// MI455X (gfx1250) — matrix operand order repaired and checked by reading, not compiled, not run on hardware
//
#include <hip/hip_runtime.h>

// Problem constants (reference: B=4, T=2048, C=1024, H=16, D=64)
#define BB 4
#define TT 2048
#define CC 1024
#define HH 16
#define DD 64
#define MM (BB * TT)   // 8192 rows in flattened [B*T, C]

typedef __attribute__((ext_vector_type(16))) __bf16          v16bf;
typedef __attribute__((ext_vector_type(8)))  float           v8f;
typedef __attribute__((ext_vector_type(8)))  unsigned short  ush8;   // 16-byte chunk
typedef int v4i_ __attribute__((vector_size(16)));                   // 128-bit block

union FragU { v16bf v; ush8 h[2]; };

static __device__ inline v16bf make_frag(ush8 lo, ush8 hi) {
    FragU u; u.h[0] = lo; u.h[1] = hi; return u.v;
}
static __device__ inline ush8 ld8(const unsigned short* p) {
    return *reinterpret_cast<const ush8*>(p);
}
static __device__ inline unsigned short f2bf(float f) {
    unsigned int u = __float_as_uint(f);
    unsigned int r = (u + 0x7FFFu + ((u >> 16) & 1u)) >> 16;
    return (unsigned short)r;
}
static __device__ inline v8f vzero8() {
    v8f z = {0.f, 0.f, 0.f, 0.f, 0.f, 0.f, 0.f, 0.f};
    return z;
}

// ---------------------------------------------------------------------------
// gfx1250 async global->LDS copy (ASYNCcnt path), with portable fallback.
// Builtin signature (from hipcc diagnostic): params are pointers to 128-bit
// int vectors in AS1 (global / __device__) and AS3 (LDS), + imm offset, cpol.
// ---------------------------------------------------------------------------
#if __has_builtin(__builtin_amdgcn_global_load_async_to_lds_b128) && \
    __has_builtin(__builtin_amdgcn_s_wait_asynccnt)
#define USE_ASYNC_LDS 1
#else
#define USE_ASYNC_LDS 0
#endif

static __device__ inline void copy16_g2l(const unsigned short* g, unsigned short* l) {
#if USE_ASYNC_LDS
    __builtin_amdgcn_global_load_async_to_lds_b128(
        (__attribute__((address_space(1))) v4i_*)g,
        (__attribute__((address_space(3))) v4i_*)l, 0, 0);
#else
    *reinterpret_cast<ush8*>(l) = *reinterpret_cast<const ush8*>(g);
#endif
}
static __device__ inline void wait_async_copies() {
#if USE_ASYNC_LDS
    __builtin_amdgcn_s_wait_asynccnt(0);
#endif
}

// ---------------------------------------------------------------------------
// Stage 1: fp32 -> bf16 conversion (grid-stride)
// ---------------------------------------------------------------------------
__global__ void cvt_f32_bf16(const float* __restrict__ in,
                             unsigned short* __restrict__ out, int n) {
    for (int i = blockIdx.x * blockDim.x + threadIdx.x; i < n;
         i += gridDim.x * blockDim.x)
        out[i] = f2bf(in[i]);
}

// ---------------------------------------------------------------------------
// Stage 2/4: Y = X @ W^T + bias.  Block-cooperative WMMA GEMM:
//   block = 256 threads = 8 waves, 128x128 output tile,
//   k-chunks of 32 staged in LDS (A 128x32, B 128x32),
//   each wave computes a 32x64 subtile (2 M-frags x 4 N-frags, 8 WMMAs/chunk).
// MODE 0: bf16 out, [B,H,T,D] scatter (Q / K)
// MODE 1: bf16 out, [B,H,D,T] scatter (V transposed)
// MODE 2: fp32 out, row-major [M, C]   (final projection -> d_out)
// ---------------------------------------------------------------------------
template <int MODE>
__global__ __launch_bounds__(256) void gemm_xwT(
    const unsigned short* __restrict__ A,
    const unsigned short* __restrict__ W,
    const float* __restrict__ bias,
    unsigned short* __restrict__ outB,
    float* __restrict__ outF) {

    __shared__ unsigned short As[128][40];   // 128 rows x 32 k (+16B pad)
    __shared__ unsigned short Bs[128][40];

    const int tid   = threadIdx.x;
    const int lane  = tid & 31;
    const int wid   = tid >> 5;
    const int wid_m = wid >> 1;              // 0..3 -> 32-row band
    const int wid_n = wid & 1;               // 0..1 -> 64-col band
    const int m0b = (blockIdx.x >> 3) * 128;
    const int n0b = (blockIdx.x & 7) * 128;
    const int r  = lane & 15;
    const int hi = lane >> 4;

    v8f acc[2][4];
#pragma unroll
    for (int am = 0; am < 2; ++am)
#pragma unroll
        for (int t = 0; t < 4; ++t) acc[am][t] = vzero8();

    for (int k = 0; k < CC; k += 32) {
        __syncthreads();                     // previous-iteration reads done
        // ---- cooperative staging: 512 16B chunks per tile, 2 per thread ----
#pragma unroll
        for (int i = 0; i < 2; ++i) {
            int c = tid * 2 + i;             // 0..511
            int row = c >> 2, col = (c & 3) * 8;
            copy16_g2l(A + (size_t)(m0b + row) * CC + k + col, &As[row][col]);
            copy16_g2l(W + (size_t)(n0b + row) * CC + k + col, &Bs[row][col]);
        }
        wait_async_copies();
        __syncthreads();

        // ---- fragments from LDS ----
        v16bf af[2];
#pragma unroll
        for (int am = 0; am < 2; ++am) {
            const unsigned short* ar = &As[wid_m * 32 + am * 16 + r][0];
            af[am] = make_frag(ld8(ar + hi * 16), ld8(ar + hi * 16 + 8));
        }
#pragma unroll
        for (int t = 0; t < 4; ++t) {
            const unsigned short* br = &Bs[wid_n * 64 + t * 16 + r][hi * 16];
            v16bf bf = make_frag(ld8(br), ld8(br + 8));
#pragma unroll
            for (int am = 0; am < 2; ++am)
                acc[am][t] = __builtin_amdgcn_wmma_f32_16x16x32_bf16(
                    false, af[am], false, bf, (short)0, acc[am][t], false, false);
        }
    }

#pragma unroll
    for (int am = 0; am < 2; ++am)
#pragma unroll
        for (int t = 0; t < 4; ++t)
#pragma unroll
            for (int j = 0; j < 8; ++j) {
                int m = m0b + wid_m * 32 + am * 16 + j + 8 * hi;
                int n = n0b + wid_n * 64 + t * 16 + r;
                float val = acc[am][t][j] + bias[n];
                if (MODE == 2) {
                    outF[(size_t)m * CC + n] = val;
                } else {
                    int b = m >> 11, tq = m & (TT - 1);
                    int h = n >> 6,  d  = n & (DD - 1);
                    size_t dst = (MODE == 0)
                        ? ((size_t)(b * HH + h) * TT + tq) * DD + d
                        : ((size_t)(b * HH + h) * DD + d) * TT + tq;
                    outB[dst] = f2bf(val);
                }
            }
}

// ---------------------------------------------------------------------------
// Stage 3: causal flash attention.  One wave == one 16-query tile of one
// (b,h); keys processed in 64-wide chunks (4 S-frags -> 16 WMMAs/chunk).
// Q,K: [B,H,T,D] bf16.  Vt: [B,H,D,T] bf16.  Y out: [B*T, C] bf16.
// ---------------------------------------------------------------------------
__global__ __launch_bounds__(256) void attn_flash(
    const unsigned short* __restrict__ Q,
    const unsigned short* __restrict__ K,
    const unsigned short* __restrict__ Vt,
    unsigned short* __restrict__ Y) {

    __shared__ unsigned short Plds[8][16][72];   // per-wave 16x64 P tile (+pad)

    const int lane = threadIdx.x & 31;
    const int wid  = threadIdx.x >> 5;
    const int wt   = blockIdx.x * 8 + wid;
    const int qTiles = TT / 16;                  // 128
    const int bh  = wt / qTiles;
    const int m0  = (wt % qTiles) * 16;
    const int b   = bh / HH, h = bh % HH;
    const int r   = lane & 15;
    const int hi  = lane >> 4;
    const float scale = 0.125f;                  // 1/sqrt(64)
    const float NEG = -3.0e38f;

    // Q fragments (K-dim 64 -> two 16x32 A-frags), loaded once.
    const unsigned short* qrow = Q + ((size_t)bh * TT + m0 + r) * DD;
    v16bf a0 = make_frag(ld8(qrow + hi * 16),      ld8(qrow + 8 + hi * 16));
    v16bf a1 = make_frag(ld8(qrow + 32 + hi * 16), ld8(qrow + 40 + hi * 16));

    v8f o[4];
#pragma unroll
    for (int t = 0; t < 4; ++t) o[t] = vzero8();
    float mrow[8], lrow[8];
#pragma unroll
    for (int j = 0; j < 8; ++j) { mrow[j] = NEG; lrow[j] = 0.f; }

    const int nChunks = ((m0 + 15) >> 6) + 1;    // 64-key chunks, causal bound
    for (int c = 0; c < nChunks; ++c) {
        const int c0 = c * 64;

        // ---- S = Q @ K^T, four 16x16 N-tiles ----
        v8f s[4];
#pragma unroll
        for (int f = 0; f < 4; ++f) {
            const unsigned short* krow =
                K + ((size_t)bh * TT + c0 + f * 16 + r) * DD + hi * 16;
            v16bf bk0 = make_frag(ld8(krow),      ld8(krow + 8));
            v16bf bk1 = make_frag(ld8(krow + 32), ld8(krow + 40));
            v8f sf = __builtin_amdgcn_wmma_f32_16x16x32_bf16(
                false, a0, false, bk0, (short)0, vzero8(), false, false);
            sf = __builtin_amdgcn_wmma_f32_16x16x32_bf16(
                false, a1, false, bk1, (short)0, sf, false, false);
            s[f] = sf;
        }

        // ---- scale + causal mask ----
#pragma unroll
        for (int f = 0; f < 4; ++f) {
            int key = c0 + f * 16 + r;
#pragma unroll
            for (int j = 0; j < 8; ++j) {
                int q = m0 + j + 8 * hi;
                float v = s[f][j] * scale;
                s[f][j] = (key > q) ? NEG : v;
            }
        }

        // ---- online softmax (rows live across 16 lanes of a half-wave) ----
#pragma unroll
        for (int j = 0; j < 8; ++j) {
            float rm = fmaxf(fmaxf(s[0][j], s[1][j]), fmaxf(s[2][j], s[3][j]));
#pragma unroll
            for (int off = 1; off < 16; off <<= 1)
                rm = fmaxf(rm, __shfl_xor(rm, off, 16));
            float mn = fmaxf(mrow[j], rm);
            float alpha = __expf(mrow[j] - mn);
            float rs = 0.f;
#pragma unroll
            for (int f = 0; f < 4; ++f) {
                float p = __expf(s[f][j] - mn);
                s[f][j] = p;
                rs += p;
            }
#pragma unroll
            for (int off = 1; off < 16; off <<= 1)
                rs += __shfl_xor(rs, off, 16);
            lrow[j] = lrow[j] * alpha + rs;
            mrow[j] = mn;
#pragma unroll
            for (int t = 0; t < 4; ++t) o[t][j] *= alpha;
        }

        // ---- P (D-layout) -> LDS row-major -> A-layout fragments ----
#pragma unroll
        for (int f = 0; f < 4; ++f)
#pragma unroll
            for (int j = 0; j < 8; ++j)
                Plds[wid][j + 8 * hi][f * 16 + r] = f2bf(s[f][j]);

        const unsigned short* prow = &Plds[wid][r][0];
        v16bf pa0 = make_frag(ld8(prow + hi * 16),      ld8(prow + 8 + hi * 16));
        v16bf pa1 = make_frag(ld8(prow + 32 + hi * 16), ld8(prow + 40 + hi * 16));

        // ---- O += P @ V  (Vt gives contiguous B-fragment columns) ----
#pragma unroll
        for (int t = 0; t < 4; ++t) {
            const unsigned short* vrow =
                Vt + ((size_t)bh * DD + t * 16 + r) * TT + c0 + hi * 16;
            v16bf bv0 = make_frag(ld8(vrow),      ld8(vrow + 8));
            v16bf bv1 = make_frag(ld8(vrow + 32), ld8(vrow + 40));
            o[t] = __builtin_amdgcn_wmma_f32_16x16x32_bf16(
                false, pa0, false, bv0, (short)0, o[t], false, false);
            o[t] = __builtin_amdgcn_wmma_f32_16x16x32_bf16(
                false, pa1, false, bv1, (short)0, o[t], false, false);
        }
    }

    // ---- normalize + store Y[b*T + q, h*64 + d] (bf16) ----
#pragma unroll
    for (int j = 0; j < 8; ++j) {
        float inv = 1.0f / lrow[j];
        int q = m0 + j + 8 * hi;
        unsigned short* yrow = Y + ((size_t)b * TT + q) * CC + h * DD;
#pragma unroll
        for (int t = 0; t < 4; ++t)
            yrow[t * 16 + r] = f2bf(o[t][j] * inv);
    }
}

// ---------------------------------------------------------------------------
// Launch: inputs are x, Wq, bq, Wk, bk, Wv, bv, Wp, bp (all fp32).
// ---------------------------------------------------------------------------
extern "C" void kernel_launch(void* const* d_in, const int* in_sizes, int n_in,
                              void* d_out, int out_size, void* d_ws, size_t ws_size,
                              hipStream_t stream) {
    const float* x  = (const float*)d_in[0];
    const float* Wq = (const float*)d_in[1];
    const float* bq = (const float*)d_in[2];
    const float* Wk = (const float*)d_in[3];
    const float* bk = (const float*)d_in[4];
    const float* Wv = (const float*)d_in[5];
    const float* bv = (const float*)d_in[6];
    const float* Wp = (const float*)d_in[7];
    const float* bp = (const float*)d_in[8];
    float* out = (float*)d_out;

    // Workspace layout (bf16 = ushort), ~88 MB total.
    char* ws = (char*)d_ws;
    const size_t nX = (size_t)MM * CC;           // 8M elements
    const size_t nW = (size_t)CC * CC;           // 1M elements
    unsigned short* Xb  = (unsigned short*)(ws);      size_t o1 = nX * 2;
    unsigned short* Wqb = (unsigned short*)(ws + o1); o1 += nW * 2;
    unsigned short* Wkb = (unsigned short*)(ws + o1); o1 += nW * 2;
    unsigned short* Wvb = (unsigned short*)(ws + o1); o1 += nW * 2;
    unsigned short* Wpb = (unsigned short*)(ws + o1); o1 += nW * 2;
    unsigned short* Qb  = (unsigned short*)(ws + o1); o1 += nX * 2;
    unsigned short* Kb  = (unsigned short*)(ws + o1); o1 += nX * 2;
    unsigned short* Vtb = (unsigned short*)(ws + o1); o1 += nX * 2;
    unsigned short* Yb  = (unsigned short*)(ws + o1);

    // Stage 1: conversions
    cvt_f32_bf16<<<2048, 256, 0, stream>>>(x,  Xb,  (int)nX);
    cvt_f32_bf16<<<1024, 256, 0, stream>>>(Wq, Wqb, (int)nW);
    cvt_f32_bf16<<<1024, 256, 0, stream>>>(Wk, Wkb, (int)nW);
    cvt_f32_bf16<<<1024, 256, 0, stream>>>(Wv, Wvb, (int)nW);
    cvt_f32_bf16<<<1024, 256, 0, stream>>>(Wp, Wpb, (int)nW);

    // Stage 2: projections — (8192/128)*(1024/128) = 512 blocks of 128x128
    dim3 gGemm(512), blk(256);
    gemm_xwT<0><<<gGemm, blk, 0, stream>>>(Xb, Wqb, bq, Qb,  nullptr);
    gemm_xwT<0><<<gGemm, blk, 0, stream>>>(Xb, Wkb, bk, Kb,  nullptr);
    gemm_xwT<1><<<gGemm, blk, 0, stream>>>(Xb, Wvb, bv, Vtb, nullptr);

    // Stage 3: attention — B*H*(T/16) = 8192 wave tiles / 8 = 1024 blocks
    attn_flash<<<1024, blk, 0, stream>>>(Qb, Kb, Vtb, Yb);

    // Stage 4: output projection (fp32 straight to d_out)
    gemm_xwT<2><<<gGemm, blk, 0, stream>>>(Yb, Wpb, bp, nullptr, out);
}
